// Set2Seq2Seq_60378650247666
// MI455X (gfx1250) — matrix operand order repaired and checked by reading, not compiled, not run on hardware
//
#include <hip/hip_runtime.h>
#include <hip/hip_bf16.h>

// Problem constants (fixed by the reference)
#define BSZ 512
#define WLN 50
#define TLN 50
#define HD  1024
#define VC  1024
#define MVD 64
#define LD  30
// SOS_INDEX = 0, EOS_INDEX = 1

// Register-blocking factors for the main GEMM (per-wave macro tile 32M x 64N)
#define MT 2
#define NT 4

typedef __attribute__((ext_vector_type(16))) __bf16 v16bf;
typedef __attribute__((ext_vector_type(8)))  float  v8f;

__device__ __forceinline__ float sigmoidf_(float x) { return 1.0f / (1.0f + __expf(-x)); }

union FragU { v16bf v; uint4 u[2]; };

// A-matrix 16x32 bf16 fragment per CDNA5 ISA layout:
// lanes 0-15 : M=lane,     elems 0-7 -> K=k0+0..7,   elems 8-15 -> K=k0+16..23
// lanes 16-31: M=lane-16,  elems 0-7 -> K=k0+8..15,  elems 8-15 -> K=k0+24..31
__device__ __forceinline__ v16bf load_A_frag(const __bf16* __restrict__ A, int lda,
                                             int m0, int k0, int lane) {
  const int half = lane >> 4;
  const int mr   = lane & 15;
  const __bf16* p = A + (size_t)(m0 + mr) * lda + k0 + 8 * half;
  FragU f;
  f.u[0] = *reinterpret_cast<const uint4*>(p);        // 8 bf16
  f.u[1] = *reinterpret_cast<const uint4*>(p + 16);   // 8 bf16
  return f.v;
}

// B-matrix 32x16 bf16 fragment; weights stored row-major (N x K) so B(k,n)=W[n,k].
// lanes 0-15 : N=lane,     elems 0-7 -> K=k0+0..7,   elems 8-15 -> K=k0+16..23
// lanes 16-31: N=lane-16,  elems 0-7 -> K=k0+8..15,  elems 8-15 -> K=k0+24..31
__device__ __forceinline__ v16bf load_B_frag(const __bf16* __restrict__ W, int ldk,
                                             int n0, int k0, int lane) {
  const int half = lane >> 4;
  const int nr   = lane & 15;
  const __bf16* p = W + (size_t)(n0 + nr) * ldk + k0 + 8 * half;
  FragU f;
  f.u[0] = *reinterpret_cast<const uint4*>(p);
  f.u[1] = *reinterpret_cast<const uint4*>(p + 16);
  return f.v;
}

// ---------------------------------------------------------------------------
// Register-blocked GEMM: C[M x N] = A1@W1^T (+ A2@W2^T) + bias1 + bias2 (+Cadd)
// Wave macro-tile: (16*MT) x (16*NT) = 32 x 64. Block: 8 waves (2M x 4N) ->
// block tile 64M x 256N. Requires N % 256 == 0, M % 64 == 0.
// Per K=32 step: MT+NT frag loads, MT*NT WMMAs -> 1.5 b128/WMMA.
// ---------------------------------------------------------------------------
__global__ __launch_bounds__(256) void wmma_gemm_blk_k(
    const __bf16* __restrict__ A1, const __bf16* __restrict__ W1, int K1,
    const __bf16* __restrict__ A2, const __bf16* __restrict__ W2, int K2,
    const float* __restrict__ bias1, const float* __restrict__ bias2,
    const float* __restrict__ Cadd,
    float* __restrict__ C, int N) {
  const int lane = threadIdx.x & 31;
  const int wave = threadIdx.x >> 5;
  const int wx = wave & 3;   // 4 waves along N
  const int wy = wave >> 2;  // 2 waves along M
  const int n0 = blockIdx.x * (4 * 16 * NT) + wx * (16 * NT);
  const int m0 = blockIdx.y * (2 * 16 * MT) + wy * (16 * MT);

  const v8f vzero = {0.f, 0.f, 0.f, 0.f, 0.f, 0.f, 0.f, 0.f};
  v8f acc[MT][NT];
#pragma unroll
  for (int i = 0; i < MT; ++i)
#pragma unroll
    for (int j = 0; j < NT; ++j) acc[i][j] = vzero;

  for (int k0 = 0; k0 < K1; k0 += 32) {
    v16bf a[MT], b[NT];
#pragma unroll
    for (int i = 0; i < MT; ++i) a[i] = load_A_frag(A1, K1, m0 + 16 * i, k0, lane);
#pragma unroll
    for (int j = 0; j < NT; ++j) b[j] = load_B_frag(W1, K1, n0 + 16 * j, k0, lane);
    if (k0 + 32 < K1) {  // pull next K-chunk of the weight stream toward the WGP
      const int half = lane >> 4, nr = lane & 15;
      __builtin_prefetch(W1 + (size_t)(n0 + nr) * K1 + (k0 + 32) + 16 * half, 0, 1);
    }
#pragma unroll
    for (int i = 0; i < MT; ++i)
#pragma unroll
      for (int j = 0; j < NT; ++j)
        acc[i][j] = __builtin_amdgcn_wmma_f32_16x16x32_bf16(
            false, a[i], false, b[j], (short)0, acc[i][j], false, false);
  }
  if (A2) {
    for (int k0 = 0; k0 < K2; k0 += 32) {
      v16bf a[MT], b[NT];
#pragma unroll
      for (int i = 0; i < MT; ++i) a[i] = load_A_frag(A2, K2, m0 + 16 * i, k0, lane);
#pragma unroll
      for (int j = 0; j < NT; ++j) b[j] = load_B_frag(W2, K2, n0 + 16 * j, k0, lane);
      if (k0 + 32 < K2) {
        const int half = lane >> 4, nr = lane & 15;
        __builtin_prefetch(W2 + (size_t)(n0 + nr) * K2 + (k0 + 32) + 16 * half, 0, 1);
      }
#pragma unroll
      for (int i = 0; i < MT; ++i)
#pragma unroll
        for (int j = 0; j < NT; ++j)
          acc[i][j] = __builtin_amdgcn_wmma_f32_16x16x32_bf16(
              false, a[i], false, b[j], (short)0, acc[i][j], false, false);
    }
  }

  // C/D layout: lane 0-15: N=lane, VGPR r -> M=r ; lane 16-31: N=lane-16, M=r+8
  const int half = lane >> 4;
  const int nr   = lane & 15;
#pragma unroll
  for (int j = 0; j < NT; ++j) {
    int col = n0 + 16 * j + nr;
    float bsum = 0.f;
    if (bias1) bsum += bias1[col];
    if (bias2) bsum += bias2[col];
#pragma unroll
    for (int i = 0; i < MT; ++i) {
#pragma unroll
      for (int r = 0; r < 8; ++r) {
        int row = m0 + 16 * i + r + 8 * half;
        float v = acc[i][j][r] + bsum;
        if (Cadd) v += Cadd[(size_t)row * N + col];
        C[(size_t)row * N + col] = v;
      }
    }
  }
}

// Simple wave-per-16x16-tile GEMM for small N (generator output projection N=64)
__global__ __launch_bounds__(256) void wmma_gemm_k(
    const __bf16* __restrict__ A1, const __bf16* __restrict__ W1, int K1,
    const float* __restrict__ bias1, float* __restrict__ C, int N) {
  const int lane = threadIdx.x & 31;
  const int wave = threadIdx.x >> 5;
  const int n0 = (blockIdx.x * 8 + wave) * 16;
  const int m0 = blockIdx.y * 16;
  if (n0 >= N) return;  // wave-uniform

  v8f acc = {0.f, 0.f, 0.f, 0.f, 0.f, 0.f, 0.f, 0.f};
  for (int k0 = 0; k0 < K1; k0 += 32) {
    v16bf a = load_A_frag(A1, K1, m0, k0, lane);
    v16bf b = load_B_frag(W1, K1, n0, k0, lane);
    acc = __builtin_amdgcn_wmma_f32_16x16x32_bf16(false, a, false, b,
                                                  (short)0, acc, false, false);
  }
  const int col  = n0 + (lane & 15);
  const int half = lane >> 4;
  float bsum = bias1 ? bias1[col] : 0.f;
#pragma unroll
  for (int r = 0; r < 8; ++r) {
    int row = m0 + r + 8 * half;
    C[(size_t)row * N + col] = acc[r] + bsum;
  }
}

// LSTM cell: gates G[B x 4H] split [i,f,g,o]; optional per-batch mask (menc).
__global__ __launch_bounds__(256) void lstm_cell_k(
    const float* __restrict__ G, const float* __restrict__ mask,
    float* __restrict__ h, float* __restrict__ c, __bf16* __restrict__ hbf) {
  int idx = blockIdx.x * 256 + threadIdx.x;
  if (idx >= BSZ * HD) return;
  int b = idx / HD, j = idx - b * HD;
  const float* g = G + (size_t)b * 4 * HD;
  float ig = sigmoidf_(g[j]);
  float fg = sigmoidf_(g[HD + j]);
  float gg = tanhf(g[2 * HD + j]);
  float og = sigmoidf_(g[3 * HD + j]);
  float cprev = c[idx];
  float cn = fg * cprev + ig * gg;
  float hn = og * tanhf(cn);
  if (mask) {
    float mt = mask[b];
    float hprev = h[idx];
    cn = mt * cn + (1.f - mt) * cprev;
    hn = mt * hn + (1.f - mt) * hprev;
  }
  c[idx] = cn;
  h[idx] = hn;
  hbf[idx] = (__bf16)hn;
}

// Attention + weighted sum: r[b,:] = sum_w sigmoid([h0,emb] @ attn_w + b)*mask * emb
__global__ __launch_bounds__(256) void attn_k(
    const int* __restrict__ input_var, const float* __restrict__ input_mask,
    const float* __restrict__ embedding, const float* __restrict__ attn_w,
    const float* __restrict__ attn_b, const float* __restrict__ set_h0,
    __bf16* __restrict__ r_bf) {
  int b = blockIdx.x, t = threadIdx.x;
  __shared__ float red[256];
  float p = 0.f;
#pragma unroll
  for (int i = 0; i < 4; ++i) { int hh = t + 256 * i; p += set_h0[hh] * attn_w[hh]; }
  red[t] = p; __syncthreads();
  for (int s = 128; s > 0; s >>= 1) { if (t < s) red[t] += red[t + s]; __syncthreads(); }
  float s0 = red[0] + attn_b[0];
  __syncthreads();
  float racc[4] = {0.f, 0.f, 0.f, 0.f};
  for (int w = 0; w < WLN; ++w) {
    int tok = input_var[b * WLN + w];
    const float* er = embedding + (size_t)tok * HD;
    float ev[4]; float d = 0.f;
#pragma unroll
    for (int i = 0; i < 4; ++i) { int hh = t + 256 * i; ev[i] = er[hh]; d += ev[i] * attn_w[HD + hh]; }
    red[t] = d; __syncthreads();
    for (int s = 128; s > 0; s >>= 1) { if (t < s) red[t] += red[t + s]; __syncthreads(); }
    float aw = sigmoidf_(red[0] + s0) * input_mask[w * BSZ + b];
    __syncthreads();
#pragma unroll
    for (int i = 0; i < 4; ++i) racc[i] += aw * ev[i];
  }
#pragma unroll
  for (int i = 0; i < 4; ++i) r_bf[(size_t)b * HD + t + 256 * i] = (__bf16)racc[i];
}

// softmax over MV=64, store bf16 probs for menc, record mask, update EOS mask
__global__ __launch_bounds__(64) void gen_softmax_k(
    const float* __restrict__ logits, float* __restrict__ m,
    __bf16* __restrict__ msg_bf, float* __restrict__ msg_mask) {
  int b = blockIdx.x, j = threadIdx.x;
  __shared__ float sh[64];
  float v = logits[b * MVD + j];
  sh[j] = v; __syncthreads();
  for (int s = 32; s > 0; s >>= 1) { if (j < s) sh[j] = fmaxf(sh[j], sh[j + s]); __syncthreads(); }
  float mx = sh[0]; __syncthreads();
  float e = __expf(v - mx);
  sh[j] = e; __syncthreads();
  for (int s = 32; s > 0; s >>= 1) { if (j < s) sh[j] += sh[j + s]; __syncthreads(); }
  float sum = sh[0]; __syncthreads();
  float prob = e / sum;
  msg_bf[b * MVD + j] = (__bf16)prob;
  sh[j] = prob; __syncthreads();
  if (j == 0) {
    float mo = m[b];
    msg_mask[b] = mo;
    m[b] = mo * (1.f - sh[1]);  // EOS_INDEX == 1
  }
}

__global__ void cvt_bf16_k(const float* __restrict__ s, __bf16* __restrict__ d, int n) {
  int i = blockIdx.x * 256 + threadIdx.x; if (i < n) d[i] = (__bf16)s[i];
}
__global__ void bcast_bf16_k(const float* __restrict__ s, __bf16* __restrict__ d, int K) {
  int i = blockIdx.x * 256 + threadIdx.x; if (i < BSZ * K) d[i] = (__bf16)s[i % K];
}
__global__ void bcast_f32_k(const float* __restrict__ s, float* __restrict__ d, int K) {
  int i = blockIdx.x * 256 + threadIdx.x; if (i < BSZ * K) d[i] = s[i % K];
}
__global__ void fill_f32_k(float* __restrict__ p, float v, int n) {
  int i = blockIdx.x * 256 + threadIdx.x; if (i < n) p[i] = v;
}
__global__ void dec_embed_k(const int* __restrict__ target_var,
                            const float* __restrict__ embedding, int t,
                            __bf16* __restrict__ xbf) {
  int i = blockIdx.x * 256 + threadIdx.x;
  if (i >= BSZ * HD) return;
  int b = i / HD, j = i - b * HD;
  int tok = (t == 0) ? 0 : target_var[(t - 1) * BSZ + b];  // SOS_INDEX == 0
  xbf[i] = (__bf16)embedding[(size_t)tok * HD + j];
}

extern "C" void kernel_launch(void* const* d_in, const int* in_sizes, int n_in,
                              void* d_out, int out_size, void* d_ws, size_t ws_size,
                              hipStream_t stream) {
  const int*   input_var  = (const int*)d_in[0];
  const float* input_mask = (const float*)d_in[1];
  const int*   target_var = (const int*)d_in[2];
  const float* embedding  = (const float*)d_in[4];
  const float* attn_w     = (const float*)d_in[5];
  const float* attn_b     = (const float*)d_in[6];
  const float* set_Wih    = (const float*)d_in[7];
  const float* set_Whh    = (const float*)d_in[8];
  const float* set_bih    = (const float*)d_in[9];
  const float* set_bhh    = (const float*)d_in[10];
  const float* set_h0     = (const float*)d_in[11];
  const float* set_c0     = (const float*)d_in[12];
  const float* gen_x0     = (const float*)d_in[13];
  const float* gen_Wih    = (const float*)d_in[14];
  const float* gen_Whh    = (const float*)d_in[15];
  const float* gen_bih    = (const float*)d_in[16];
  const float* gen_bhh    = (const float*)d_in[17];
  const float* gen_outW   = (const float*)d_in[18];
  const float* gen_outb   = (const float*)d_in[19];
  const float* menc_Wih   = (const float*)d_in[20];
  const float* menc_Whh   = (const float*)d_in[21];
  const float* menc_bih   = (const float*)d_in[22];
  const float* menc_bhh   = (const float*)d_in[23];
  const float* menc_h0    = (const float*)d_in[24];
  const float* menc_c0    = (const float*)d_in[25];
  const float* dec_Wih    = (const float*)d_in[26];
  const float* dec_Whh    = (const float*)d_in[27];
  const float* dec_bih    = (const float*)d_in[28];
  const float* dec_bhh    = (const float*)d_in[29];
  const float* dec_outW   = (const float*)d_in[30];
  const float* dec_outb   = (const float*)d_in[31];
  float* out = (float*)d_out;

  // ---- workspace carving (256B aligned) ----
  char* ws = (char*)d_ws;
  size_t off = 0;
  auto alloc = [&](size_t bytes) -> void* {
    void* p = ws + off;
    off = (off + bytes + 255) & ~(size_t)255;
    return p;
  };
  __bf16* wb_setWih  = (__bf16*)alloc((size_t)4 * HD * HD * 2);
  __bf16* wb_setWhh  = (__bf16*)alloc((size_t)4 * HD * HD * 2);
  __bf16* wb_genWih  = (__bf16*)alloc((size_t)4 * HD * MVD * 2);
  __bf16* wb_genWhh  = (__bf16*)alloc((size_t)4 * HD * HD * 2);
  __bf16* wb_genOutW = (__bf16*)alloc((size_t)MVD * HD * 2);
  __bf16* wb_mencWih = (__bf16*)alloc((size_t)4 * HD * MVD * 2);
  __bf16* wb_mencWhh = (__bf16*)alloc((size_t)4 * HD * HD * 2);
  __bf16* wb_decWih  = (__bf16*)alloc((size_t)4 * HD * HD * 2);
  __bf16* wb_decWhh  = (__bf16*)alloc((size_t)4 * HD * HD * 2);
  __bf16* wb_decOutW = (__bf16*)alloc((size_t)VC * HD * 2);
  float*  G        = (float*)alloc((size_t)BSZ * 4 * HD * 4);
  float*  Base     = (float*)alloc((size_t)BSZ * 4 * HD * 4);
  float*  hbuf     = (float*)alloc((size_t)BSZ * HD * 4);
  float*  cbuf     = (float*)alloc((size_t)BSZ * HD * 4);
  __bf16* hbf      = (__bf16*)alloc((size_t)BSZ * HD * 2);
  __bf16* xbf      = (__bf16*)alloc((size_t)BSZ * HD * 2);
  __bf16* rbf      = (__bf16*)alloc((size_t)BSZ * HD * 2);
  __bf16* gxbf     = (__bf16*)alloc((size_t)BSZ * MVD * 2);
  __bf16* msg      = (__bf16*)alloc((size_t)LD * BSZ * MVD * 2);
  float*  msg_mask = (float*)alloc((size_t)LD * BSZ * 4);
  float*  mbuf     = (float*)alloc((size_t)BSZ * 4);
  float*  glog     = (float*)alloc((size_t)BSZ * MVD * 4);

  auto cvt = [&](const float* s, __bf16* d, int n) {
    cvt_bf16_k<<<(n + 255) / 256, 256, 0, stream>>>(s, d, n);
  };
  // Dispatch: register-blocked kernel when N allows (N%256==0), else simple.
  auto gemm = [&](const __bf16* A1, const __bf16* W1, int K1,
                  const __bf16* A2, const __bf16* W2, int K2,
                  const float* b1, const float* b2, const float* Cadd,
                  float* C, int N) {
    if (N % 256 == 0) {
      dim3 grid(N / 256, BSZ / 64);
      wmma_gemm_blk_k<<<grid, 256, 0, stream>>>(A1, W1, K1, A2, W2, K2, b1, b2, Cadd, C, N);
    } else {
      dim3 grid((N + 127) / 128, BSZ / 16);
      wmma_gemm_k<<<grid, 256, 0, stream>>>(A1, W1, K1, b1, C, N);
    }
  };
  auto cell = [&](const float* mask) {
    lstm_cell_k<<<(BSZ * HD + 255) / 256, 256, 0, stream>>>(G, mask, hbuf, cbuf, hbf);
  };

  // ---- convert weights to bf16 (once per launch; resident in L2 thereafter) ----
  cvt(set_Wih, wb_setWih, 4 * HD * HD);
  cvt(set_Whh, wb_setWhh, 4 * HD * HD);
  cvt(gen_Wih, wb_genWih, 4 * HD * MVD);
  cvt(gen_Whh, wb_genWhh, 4 * HD * HD);
  cvt(gen_outW, wb_genOutW, MVD * HD);
  cvt(menc_Wih, wb_mencWih, 4 * HD * MVD);
  cvt(menc_Whh, wb_mencWhh, 4 * HD * HD);
  cvt(dec_Wih, wb_decWih, 4 * HD * HD);
  cvt(dec_Whh, wb_decWhh, 4 * HD * HD);
  cvt(dec_outW, wb_decOutW, VC * HD);

  // ---- encoder: attention + one LSTM cell ----
  bcast_bf16_k<<<(BSZ * HD + 255) / 256, 256, 0, stream>>>(set_h0, hbf, HD);
  bcast_f32_k<<<(BSZ * HD + 255) / 256, 256, 0, stream>>>(set_c0, cbuf, HD);
  attn_k<<<BSZ, 256, 0, stream>>>(input_var, input_mask, embedding, attn_w, attn_b, set_h0, rbf);
  gemm(rbf, wb_setWih, HD, hbf, wb_setWhh, HD, set_bih, set_bhh, nullptr, G, 4 * HD);
  cell(nullptr);

  // ---- generator loop (gx constant -> precompute Base once) ----
  bcast_bf16_k<<<(BSZ * MVD + 255) / 256, 256, 0, stream>>>(gen_x0, gxbf, MVD);
  fill_f32_k<<<(BSZ + 255) / 256, 256, 0, stream>>>(mbuf, 1.0f, BSZ);
  gemm(gxbf, wb_genWih, MVD, nullptr, nullptr, 0, gen_bih, gen_bhh, nullptr, Base, 4 * HD);
  for (int l = 0; l < LD; ++l) {
    gemm(hbf, wb_genWhh, HD, nullptr, nullptr, 0, nullptr, nullptr, Base, G, 4 * HD);
    cell(nullptr);
    gemm(hbf, wb_genOutW, HD, nullptr, nullptr, 0, gen_outb, nullptr, nullptr, glog, MVD);
    gen_softmax_k<<<BSZ, 64, 0, stream>>>(glog, mbuf, msg + (size_t)l * BSZ * MVD,
                                          msg_mask + (size_t)l * BSZ);
  }

  // ---- message encoder loop (masked state updates) ----
  bcast_bf16_k<<<(BSZ * HD + 255) / 256, 256, 0, stream>>>(menc_h0, hbf, HD);
  bcast_f32_k<<<(BSZ * HD + 255) / 256, 256, 0, stream>>>(menc_h0, hbuf, HD);
  bcast_f32_k<<<(BSZ * HD + 255) / 256, 256, 0, stream>>>(menc_c0, cbuf, HD);
  for (int l = 0; l < LD; ++l) {
    gemm(msg + (size_t)l * BSZ * MVD, wb_mencWih, MVD, hbf, wb_mencWhh, HD,
         menc_bih, menc_bhh, nullptr, G, 4 * HD);
    cell(msg_mask + (size_t)l * BSZ);
  }

  // ---- decoder loop: states carry over from menc ----
  for (int t = 0; t < TLN; ++t) {
    dec_embed_k<<<(BSZ * HD + 255) / 256, 256, 0, stream>>>(target_var, embedding, t, xbf);
    gemm(xbf, wb_decWih, HD, hbf, wb_decWhh, HD, dec_bih, dec_bhh, nullptr, G, 4 * HD);
    cell(nullptr);
    gemm(hbf, wb_decOutW, HD, nullptr, nullptr, 0, dec_outb, nullptr, nullptr,
         out + (size_t)t * BSZ * VC, VC);
  }
}
